// Conv_Mesh_9818295239460
// MI455X (gfx1250) — compile-verified, matrix operand order checked by reading, not run on hardware
//
#include <hip/hip_runtime.h>
#include <hip/hip_bf16.h>

typedef __attribute__((ext_vector_type(16))) _Float16 v16h;
typedef __attribute__((ext_vector_type(8)))  _Float16 v8h;
typedef __attribute__((ext_vector_type(8)))  float    v8f;

// ---------------------------------------------------------------------------
// Pass 1a: xu_pad[j][m] = x_pad[j] . u[m]   (row 0 = zeros; row j = x[j-1])
// LDS-tiled: 16 xu rows per block, x and u staged in LDS (bank-padded).
// ---------------------------------------------------------------------------
__global__ __launch_bounds__(256) void xu_kernel(const float* __restrict__ x,
                                                 const float* __restrict__ u,
                                                 float* __restrict__ xu, int N) {
  __shared__ float xs[16 * 132];   // padded stride: bank-conflict free
  __shared__ float us[16 * 132];
  const int tid = threadIdx.x;
  const int j0  = blockIdx.x * 16;             // xu rows j0..j0+15 of N+1 rows

#pragma unroll
  for (int t = 0; t < 2; ++t) {                // 512 float4 tasks for x
    int task = t * 256 + tid;
    int slot = task >> 5, q4 = task & 31;
    int j = j0 + slot;
    float4 v = make_float4(0.f, 0.f, 0.f, 0.f);
    if (j > 0 && j <= N) v = ((const float4*)(x + (size_t)(j - 1) * 128))[q4];
    float* d = &xs[slot * 132 + q4 * 4];
    d[0] = v.x; d[1] = v.y; d[2] = v.z; d[3] = v.w;
  }
#pragma unroll
  for (int t = 0; t < 2; ++t) {                // 512 float4 tasks for u
    int task = t * 256 + tid;
    int m = task >> 5, q4 = task & 31;
    float4 v = ((const float4*)(u + (size_t)m * 128))[q4];
    float* d = &us[m * 132 + q4 * 4];
    d[0] = v.x; d[1] = v.y; d[2] = v.z; d[3] = v.w;
  }
  __syncthreads();

  int slot = tid >> 4, m = tid & 15;
  const float* xr = &xs[slot * 132];
  const float* ur = &us[m * 132];
  float s = 0.f;
#pragma unroll 8
  for (int c = 0; c < 128; ++c) s += xr[c] * ur[c];
  int j = j0 + slot;
  if (j <= N) xu[(size_t)j * 16 + m] = s;
}

// ---------------------------------------------------------------------------
// Pass 1b: swizzle W [16][128][128] f32 -> f16 WMMA-B operand order for the
// big GEMM out[n][o] = sum_K weighted[n][K] * Wt[K][o], K = m*128 + c.
// Layout (halfs): ((kc*8 + oc)*32 + lane)*16 + i, where operand index
// i maps to kk = (i>>3)*16 + (lane>>4)*8 + (i&7); o = oc*16 + lane%16.
// ---------------------------------------------------------------------------
__global__ __launch_bounds__(256) void wswz_kernel(const float* __restrict__ W,
                                                   _Float16* __restrict__ Bswz) {
  int idx = blockIdx.x * blockDim.x + threadIdx.x;   // 16*128*128 = 262144
  if (idx >= 16 * 128 * 128) return;
  int i    = idx & 15;          // operand half index
  int lane = (idx >> 4) & 31;
  int oc   = (idx >> 9) & 7;
  int kc   = idx >> 12;         // 0..63
  int kk   = ((i >> 3) << 4) + ((lane >> 4) << 3) + (i & 7);
  int K    = kc * 32 + kk;
  int m    = K >> 7;
  int c    = K & 127;
  int o    = oc * 16 + (lane & 15);
  Bswz[idx] = (_Float16)W[((size_t)m * 128 + o) * 128 + c];
}

// ---------------------------------------------------------------------------
// Main kernel: one block (256 thr, 8 waves) per 16-node tile.
// ---------------------------------------------------------------------------
__global__ __launch_bounds__(256) void conv_main(const float* __restrict__ x,
                                                 const int* __restrict__ adj,
                                                 const float* __restrict__ xu,
                                                 const _Float16* __restrict__ Bswz,
                                                 const float* __restrict__ bvec,
                                                 const float* __restrict__ cvec,
                                                 float* __restrict__ out, int N) {
  // q in WMMA-A swizzled layout: [node][lane][8 halfs] (k = 0..15, pad hi)
  __shared__ _Float16 qswz[16 * 32 * 8];        // 8 KB
  // weighted as phase-3 A operand, deinterleaved into two 16B groups:
  // halfs index = (kc*64 + g*32 + lane)*8 + pos   (g = kk>>4, pos = kk&7,
  // lane = (kk>>3 & 1)*16 + node)  -> each ds_load_b128 has 16B lane stride.
  __shared__ _Float16 wswz[64 * 64 * 8];        // 64 KB
  __shared__ int   rowidx[16 * 16];
  __shared__ int   degcnt[16];
  __shared__ float invdeg[16];

  const int tid    = threadIdx.x;
  const int n0     = blockIdx.x * 16;
  const int wave   = tid >> 5;
  const int lane   = tid & 31;
  const int halfid = lane >> 4;
  const int lm     = lane & 15;

  if (tid < 16) degcnt[tid] = 0;
  __syncthreads();

  // ---- Phase 1: softmax q per (node, k); zero q column for padded k ----
  {
    int n  = tid >> 4, k = tid & 15;
    int ng = n0 + n;
    int a  = (ng < N) ? adj[(size_t)ng * 16 + k] : 0;
    rowidx[tid] = a;
    if (a != 0) atomicAdd(&degcnt[n], 1);

    const float* su = xu + (size_t)(ng < N ? ng + 1 : 0) * 16;
    const float* nu = xu + (size_t)a * 16;
    float lg[16];
    float mx = -1e30f;
#pragma unroll
    for (int m = 0; m < 16; ++m) {
      float v = su[m] - nu[m] + cvec[m];
      lg[m] = v;
      mx = fmaxf(mx, v);
    }
    float ssum = 0.f;
#pragma unroll
    for (int m = 0; m < 16; ++m) {
      float ev = __expf(lg[m] - mx);
      lg[m] = ev;
      ssum += ev;
    }
    // padded neighbor => zero its whole q column (kills its B row in WMMA)
    float scale = (a != 0) ? (1.f / ssum) : 0.f;
    int h = k >> 3, j = (k & 7) >> 1, e = k & 1;
#pragma unroll
    for (int m = 0; m < 16; ++m)
      qswz[((n * 32 + h * 16 + m) << 3) + (j << 1) + e] = (_Float16)(lg[m] * scale);
  }
  __syncthreads();

  if (tid < 16) {
    int d = degcnt[tid];
    invdeg[tid] = (d > 0) ? 1.f / (float)d : 0.f;
  }

  // ---- Phase 2: weighted[n] = q^T (16x16, K-pad 32) @ patches (16x128) ----
  for (int nn = wave; nn < 16; nn += 8) {
    v16h A;
    {
      v8h q8 = *(const v8h*)&qswz[(nn * 32 + lane) << 3];
#pragma unroll
      for (int i = 0; i < 8; ++i) A[i] = q8[i];
#pragma unroll
      for (int i = 8; i < 16; ++i) A[i] = (_Float16)0.f;
    }
    // hoist per-lane row base pointers (branch-free; a==0 -> x row 0,
    // contribution already zeroed via q)
    const int* ridx = &rowidx[nn * 16];
    const float* basep[8];
#pragma unroll
    for (int i = 0; i < 8; ++i) {
      int a   = ridx[(halfid << 3) + i];
      int row = (a > 0) ? (a - 1) : 0;
      basep[i] = x + (size_t)row * 128 + lm;
    }
#pragma unroll 2
    for (int cc = 0; cc < 8; ++cc) {
      int c = cc * 16 + lm;
      v16h B;
#pragma unroll
      for (int i = 0; i < 8; ++i) B[i] = (_Float16)basep[i][cc * 16];
#pragma unroll
      for (int i = 8; i < 16; ++i) B[i] = (_Float16)0.f;

      v8f C = {};
      C = __builtin_amdgcn_wmma_f32_16x16x32_f16(false, A, false, B,
                                                 (short)0, C, false, false);
      // scatter D ([m = r+8*halfid][c], f32) into phase-3 A layout as f16
#pragma unroll
      for (int r = 0; r < 8; ++r) {
        int m   = r + (halfid << 3);
        int K   = (m << 7) + c;
        int kc2 = K >> 5, kk = K & 31;
        int h2  = (kk >> 3) & 1;
        int g   = kk >> 4;
        int pos = kk & 7;
        wswz[((kc2 * 64 + g * 32 + h2 * 16 + nn) << 3) + pos] = (_Float16)C[r];
      }
    }
  }
  __syncthreads();

  // ---- Phase 3: out[16 nodes][oc*16..+16] = weighted @ Wt, K = 2048 ----
  {
    const int oc = wave;                        // 0..7 column chunks
    const v8h* ab = (const v8h*)wswz;
    v8f acc = {};
#pragma unroll 4
    for (int kc = 0; kc < 64; ++kc) {
      v8h a0 = ab[kc * 64 + lane];              // group 0: kk 0..15 region
      v8h a1 = ab[kc * 64 + 32 + lane];         // group 1: kk 16..31 region
      v16h A;
#pragma unroll
      for (int i = 0; i < 8; ++i) { A[i] = a0[i]; A[i + 8] = a1[i]; }
      v16h B = *(const v16h*)&Bswz[(size_t)((kc * 8 + oc) * 32 + lane) << 4];
      acc = __builtin_amdgcn_wmma_f32_16x16x32_f16(false, A, false, B,
                                                   (short)0, acc, false, false);
    }
    int o = oc * 16 + lm;
    float bb = bvec[o];
#pragma unroll
    for (int r = 0; r < 8; ++r) {
      int node = r + (halfid << 3);
      int ng = n0 + node;
      if (ng < N) out[(size_t)ng * 128 + o] = acc[r] * invdeg[node] + bb;
    }
  }
}

// ---------------------------------------------------------------------------
extern "C" void kernel_launch(void* const* d_in, const int* in_sizes, int n_in,
                              void* d_out, int out_size, void* d_ws, size_t ws_size,
                              hipStream_t stream) {
  const float* x    = (const float*)d_in[0];   // [N,128]
  const int*   adj  = (const int*)d_in[1];     // [N,16]
  const float* W    = (const float*)d_in[2];   // [16,128,128]
  const float* b    = (const float*)d_in[3];   // [128]
  const float* u    = (const float*)d_in[4];   // [16,128]
  const float* cvec = (const float*)d_in[5];   // [16]
  float* out = (float*)d_out;
  const int N = in_sizes[0] / 128;

  float* xu = (float*)d_ws;
  size_t xu_bytes = (size_t)(N + 1) * 16 * sizeof(float);
  size_t off = (xu_bytes + 255) & ~(size_t)255;
  _Float16* Bswz = (_Float16*)((char*)d_ws + off);

  int nrows = N + 1;                           // xu rows
  xu_kernel<<<(nrows + 15) / 16, 256, 0, stream>>>(x, u, xu, N);
  wswz_kernel<<<(16 * 128 * 128 + 255) / 256, 256, 0, stream>>>(W, Bswz);
  conv_main<<<(N + 15) / 16, 256, 0, stream>>>(x, adj, xu, Bswz, b, cvec, out, N);
}
